// EncoderLayer_2937757630987
// MI455X (gfx1250) — matrix operand order repaired and checked by reading, not compiled, not run on hardware
//
#include <hip/hip_runtime.h>
#include <hip/hip_bf16.h>

// ---------------------------------------------------------------------------
// Spatio-temporal encoder layer for MI455X (gfx1250, wave32, WMMA + TDM).
// - All matmuls: v_wmma_f32_16x16x32_bf16 (bf16 in, f32 accumulate).
// - All GEMM operands pre-staged as bf16 (x and residual outputs dual-stored,
//   weights pre-transposed to [N][K] bf16 once per launch) so every GEMM tile
//   is a contiguous 2D copy -> staged by the Tensor Data Mover
//   (tensor_load_to_lds + s_wait_tensorcnt), double-buffered.
// - Attention head-slices (128x64) fused entirely in LDS.
// ---------------------------------------------------------------------------

typedef __attribute__((ext_vector_type(16))) __bf16 v16bf;
typedef __attribute__((ext_vector_type(8)))  __bf16 v8bf;
typedef __attribute__((ext_vector_type(8)))  float  v8f;
typedef __attribute__((ext_vector_type(4)))  unsigned int u32x4;
typedef __attribute__((ext_vector_type(8)))  int i32x8;
typedef __attribute__((ext_vector_type(4)))  int i32x4;

#define D_MODEL 512
#define NHEADS  8
#define DKH     64
#define DFF     2048
#define NB      4
#define SSZ     128
#define TSZ     128
#define TOKENS  (NB * SSZ * TSZ)   // 65536 tokens of dim 512

__device__ __forceinline__ __bf16 f2bf(float f) { return (__bf16)f; }

union FragAB { v16bf v; v8bf h[2]; };
union FragC  { v8f   v; float f[8]; };

// A-matrix fragment (16x32 bf16) from row-major LDS tile [rows][ld].
// ISA layout: lane half hh: elems0-7 = K[8hh..8hh+7], elems8-15 = K[16+8hh..+7].
__device__ __forceinline__ v16bf frag_a_lds(const __bf16* base, int ld,
                                            int m0, int k0, int lane) {
  int m  = m0 + (lane & 15);
  int hh = lane >> 4;
  const __bf16* p = base + m * ld + k0;
  FragAB f;
  f.h[0] = *(const v8bf*)(p + 8 * hh);
  f.h[1] = *(const v8bf*)(p + 16 + 8 * hh);
  return f.v;
}

// B-matrix fragment (32x16 bf16) from LDS tile stored [n][ld] (k contiguous).
// ISA layout: lane half hh: elems0-7 = K[8hh..8hh+7], elems8-15 = K[16+8hh..+7].
__device__ __forceinline__ v16bf frag_b_lds(const __bf16* base, int ld,
                                            int n0, int k0, int lane) {
  int n  = n0 + (lane & 15);
  int hh = lane >> 4;
  const __bf16* p = base + n * ld + k0 + 8 * hh;
  FragAB f;
  f.h[0] = *(const v8bf*)(p);
  f.h[1] = *(const v8bf*)(p + 16);
  return f.v;
}

// ---------------------------------------------------------------------------
// TDM: issue a 2D bf16 tile load (tile_d0 x tile_d1, row stride d0_stride
// elements) from global into LDS at byte offset lds_addr.  D# per ISA §8.3/8.4:
//   group0: count=1 | lds_addr | global_addr[56:0] | type=2
//   group1: data_size=2B, tensor_dim0/1 (OOB), tile_dim0/1, tensor_dim0_stride
// This toolchain uses the 6-arg builtin (g0, g1, g2, g3, g4, cpol).
// ---------------------------------------------------------------------------
__device__ __forceinline__ void tdm_load_tile_2d(
    unsigned lds_addr, const __bf16* gptr, unsigned tensor_d0,
    unsigned tensor_d1, unsigned tile_d0, unsigned tile_d1,
    unsigned long long d0_stride) {
  unsigned long long ga = (unsigned long long)(uintptr_t)gptr;
  u32x4 g0 = {1u,                                   // count=1, user mode
              lds_addr,
              (unsigned)ga,
              (unsigned)((ga >> 32) & 0x01FFFFFFu) | 0x80000000u};  // type=2
  i32x8 g1;
  g1[0] = 0x00010000;                               // data_size = 1 -> 2 bytes
  g1[1] = (int)((tensor_d0 & 0xFFFFu) << 16);       // [47:32]=0, td0 lo
  g1[2] = (int)((tensor_d0 >> 16) | ((tensor_d1 & 0xFFFFu) << 16));
  g1[3] = (int)((tensor_d1 >> 16) | (tile_d0 << 16));
  g1[4] = (int)(tile_d1 & 0xFFFFu);                 // tile_dim2 = 0
  g1[5] = (int)(d0_stride & 0xFFFFFFFFu);
  g1[6] = (int)((d0_stride >> 32) & 0xFFFFu);       // dim1_stride = 0 (2D)
  g1[7] = 0;
  i32x4 z4 = {0, 0, 0, 0};
  i32x8 z8 = {0, 0, 0, 0, 0, 0, 0, 0};
  __builtin_amdgcn_tensor_load_to_lds(g0, g1, z4, z4, z8, 0);
}

// ---------------------------------------------------------------------------
// GEMM: C[M,N] = op(A[M,K](bf16) * Bt[N,K](bf16) + bias (+resid fp32)).
// Block tile 128x128, BK=32, 8 waves (4x2), wave = 32x64 (2x4 fragments).
// A/B tiles staged by TDM, double-buffered, synced with TENSORcnt.
// OUTMODE: 0 = bf16 only, 1 = f32 only, 2 = both (f32 + bf16 shadow).
// ---------------------------------------------------------------------------
template <bool RELU, int OUTMODE, bool HASRES>
__global__ __launch_bounds__(256) void gemm_tdm_wmma(
    const __bf16* __restrict__ A, const __bf16* __restrict__ Bt,
    const float* __restrict__ bias, const float* __restrict__ resid,
    float* __restrict__ Cf, __bf16* __restrict__ Cb, int M, int N, int K) {
  __shared__ __bf16 As[2][128 * 32];   // [m][k], 2 x 8 KB
  __shared__ __bf16 Bs[2][128 * 32];   // [n][k], 2 x 8 KB

  const int tid  = threadIdx.x;
  const int lane = tid & 31;
  const int wid  = tid >> 5;
  const int bm   = blockIdx.x * 128;
  const int bn   = blockIdx.y * 128;
  const int wm0  = (wid & 3) * 32;
  const int wn0  = (wid >> 2) * 64;

  FragC acc[2][4];
  v8f zero = {};
#pragma unroll
  for (int i = 0; i < 2; ++i)
#pragma unroll
    for (int j = 0; j < 4; ++j) acc[i][j].v = zero;

  const int nk = K >> 5;   // BK = 32

  auto issue_pair = [&](int buf, int t) {
    unsigned a_lds = (unsigned)(uintptr_t)&As[buf][0];
    unsigned b_lds = (unsigned)(uintptr_t)&Bs[buf][0];
    tdm_load_tile_2d(a_lds, A + (size_t)bm * K + t * 32,
                     (unsigned)K, (unsigned)M, 32u, 128u,
                     (unsigned long long)K);
    tdm_load_tile_2d(b_lds, Bt + (size_t)bn * K + t * 32,
                     (unsigned)K, (unsigned)N, 32u, 128u,
                     (unsigned long long)K);
  };

  if (wid == 0) issue_pair(0, 0);

  for (int t = 0; t < nk; ++t) {
    const int buf = t & 1;
    if (wid == 0) {
      if (t + 1 < nk) {
        issue_pair(buf ^ 1, t + 1);                 // prefetch next tile pair
        __builtin_amdgcn_s_wait_tensorcnt(2);       // current pair landed
      } else {
        __builtin_amdgcn_s_wait_tensorcnt(0);
      }
    }
    __syncthreads();   // publish TDM-written LDS to all 8 waves

#pragma unroll
    for (int mt = 0; mt < 2; ++mt) {
      v16bf a = frag_a_lds(&As[buf][0], 32, wm0 + mt * 16, 0, lane);
#pragma unroll
      for (int nt = 0; nt < 4; ++nt) {
        v16bf b = frag_b_lds(&Bs[buf][0], 32, wn0 + nt * 16, 0, lane);
        acc[mt][nt].v = __builtin_amdgcn_wmma_f32_16x16x32_bf16(
            false, a, false, b, (short)0, acc[mt][nt].v, false, false);
      }
    }
    __syncthreads();   // done reading buf before TDM rewrites it
  }

  // ---- epilogue: bias / relu / residual / store (f32 and/or bf16) ----
#pragma unroll
  for (int mt = 0; mt < 2; ++mt)
#pragma unroll
    for (int nt = 0; nt < 4; ++nt) {
      int col = bn + wn0 + nt * 16 + (lane & 15);
      float bv = bias[col];
#pragma unroll
      for (int r = 0; r < 8; ++r) {
        int row = bm + wm0 + mt * 16 + r + 8 * (lane >> 4);
        float v = acc[mt][nt].f[r] + bv;
        if (RELU) v = fmaxf(v, 0.0f);
        size_t idx = (size_t)row * N + col;
        if (HASRES) v += resid[idx];
        if (OUTMODE == 0 || OUTMODE == 2) Cb[idx] = f2bf(v);
        if (OUTMODE == 1 || OUTMODE == 2) Cf[idx] = v;
      }
    }
}

// ---------------------------------------------------------------------------
// Fused attention over one (n, h, slot) 128x64 head slice.
// mode 0: time attention  (fixed n,h,s; rows = t, stride D, causal mask)
// mode 1: space attention (fixed n,h,t; rows = s, stride T*D, no mask)
// Whole slice in LDS: S = QK^T/8 -> softmax -> O = P V.  144 KB dyn LDS.
// ---------------------------------------------------------------------------
__global__ __launch_bounds__(256) void attn_wmma_kernel(
    const __bf16* __restrict__ Q, const __bf16* __restrict__ K,
    const __bf16* __restrict__ V, __bf16* __restrict__ O, int mode) {
  extern __shared__ __bf16 smem[];
  __bf16* Qs = smem;               // [128][64]   16 KB
  __bf16* Ks = Qs + 128 * 64;      // [128][64]   16 KB (B tile for QK^T)
  __bf16* Vt = Ks + 128 * 64;      // [64][128]   16 KB (V transposed: [dk][u])
  float*  Ss = (float*)(Vt + 64 * 128);    // [128][128] 64 KB scores
  __bf16* Ps = (__bf16*)(Ss + 128 * 128);  // [128][128] 32 KB probabilities

  const int tid  = threadIdx.x;
  const int lane = tid & 31;
  const int wid  = tid >> 5;

  size_t base, rstride;
  const int g = blockIdx.x;
  if (mode == 0) {  // g = ((n*H)+h)*S + s
    int s = g % SSZ;
    int nh = g / SSZ;
    int h = nh % NHEADS, n = nh / NHEADS;
    base = ((size_t)(n * SSZ + s) * TSZ) * D_MODEL + (size_t)h * DKH;
    rstride = D_MODEL;
  } else {          // g = ((n*H)+h)*T + t
    int t = g % TSZ;
    int nh = g / TSZ;
    int h = nh % NHEADS, n = nh / NHEADS;
    base = ((size_t)n * SSZ * TSZ + t) * D_MODEL + (size_t)h * DKH;
    rstride = (size_t)TSZ * D_MODEL;
  }

  {  // ---- load Q,K,V slice (strided rows, contiguous dk) ----
    int r  = tid >> 1;
    int c0 = (tid & 1) * 32;
    size_t off = base + (size_t)r * rstride + c0;
#pragma unroll
    for (int i = 0; i < 4; ++i) {
      v8bf q = *(const v8bf*)(Q + off + 8 * i);
      v8bf k = *(const v8bf*)(K + off + 8 * i);
      v8bf v = *(const v8bf*)(V + off + 8 * i);
      *(v8bf*)(Qs + r * 64 + c0 + 8 * i) = q;
      *(v8bf*)(Ks + r * 64 + c0 + 8 * i) = k;
#pragma unroll
      for (int j = 0; j < 8; ++j) Vt[(c0 + 8 * i + j) * 128 + r] = v[j];
    }
  }
  __syncthreads();

  {  // ---- scores: S[128][128] = Q(128x64) * K^T, scale + causal mask ----
    int m0 = (wid & 3) * 32;
    int u0 = (wid >> 2) * 64;
    FragC acc[2][4];
    v8f zero = {};
#pragma unroll
    for (int i = 0; i < 2; ++i)
#pragma unroll
      for (int j = 0; j < 4; ++j) acc[i][j].v = zero;

#pragma unroll
    for (int kk = 0; kk < 64; kk += 32)
#pragma unroll
      for (int mt = 0; mt < 2; ++mt) {
        v16bf a = frag_a_lds(Qs, 64, m0 + mt * 16, kk, lane);
#pragma unroll
        for (int ut = 0; ut < 4; ++ut) {
          v16bf b = frag_b_lds(Ks, 64, u0 + ut * 16, kk, lane);
          acc[mt][ut].v = __builtin_amdgcn_wmma_f32_16x16x32_bf16(
              false, a, false, b, (short)0, acc[mt][ut].v, false, false);
        }
      }

    const float scale = 0.125f;  // 1/sqrt(64)
#pragma unroll
    for (int mt = 0; mt < 2; ++mt)
#pragma unroll
      for (int ut = 0; ut < 4; ++ut) {
        int u = u0 + ut * 16 + (lane & 15);
#pragma unroll
        for (int r = 0; r < 8; ++r) {
          int row = m0 + mt * 16 + r + 8 * (lane >> 4);
          float v = acc[mt][ut].f[r] * scale;
          if (mode == 0 && u > row) v = -1e30f;  // causal
          Ss[row * 128 + u] = v;
        }
      }
  }
  __syncthreads();

  if (tid < 128) {  // ---- row softmax ----
    float mx = -1e30f;
    for (int j = 0; j < 128; ++j) mx = fmaxf(mx, Ss[tid * 128 + j]);
    float sum = 0.0f;
    for (int j = 0; j < 128; ++j) {
      float e = __expf(Ss[tid * 128 + j] - mx);
      sum += e;
      Ss[tid * 128 + j] = e;
    }
    float inv = 1.0f / sum;
    for (int j = 0; j < 128; ++j) Ps[tid * 128 + j] = f2bf(Ss[tid * 128 + j] * inv);
  }
  __syncthreads();

  {  // ---- O(128x64) = P(128x128) * V(128x64), write back strided ----
    int m0 = (wid & 3) * 32;
    int n0 = (wid >> 2) * 32;
    FragC acc[2][2];
    v8f zero = {};
#pragma unroll
    for (int i = 0; i < 2; ++i)
#pragma unroll
      for (int j = 0; j < 2; ++j) acc[i][j].v = zero;

#pragma unroll
    for (int kk = 0; kk < 128; kk += 32)
#pragma unroll
      for (int mt = 0; mt < 2; ++mt) {
        v16bf a = frag_a_lds(Ps, 128, m0 + mt * 16, kk, lane);
#pragma unroll
        for (int nt = 0; nt < 2; ++nt) {
          v16bf b = frag_b_lds(Vt, 128, n0 + nt * 16, kk, lane);
          acc[mt][nt].v = __builtin_amdgcn_wmma_f32_16x16x32_bf16(
              false, a, false, b, (short)0, acc[mt][nt].v, false, false);
        }
      }

#pragma unroll
    for (int mt = 0; mt < 2; ++mt)
#pragma unroll
      for (int nt = 0; nt < 2; ++nt) {
        int col = n0 + nt * 16 + (lane & 15);
#pragma unroll
        for (int r = 0; r < 8; ++r) {
          int row = m0 + mt * 16 + r + 8 * (lane >> 4);
          O[base + (size_t)row * rstride + col] = f2bf(acc[mt][nt].f[r]);
        }
      }
  }
}

// ---------------------------------------------------------------------------
// Prep kernels (run once per launch, tiny vs. main work):
//  - cvt_f32_bf16: x -> bf16 activation shadow
//  - wtrans: weight [K][N] fp32 -> [N][K] bf16 (LDS 32x32 tile transpose)
// ---------------------------------------------------------------------------
__global__ __launch_bounds__(256) void cvt_f32_bf16(
    const float* __restrict__ in, __bf16* __restrict__ out) {
  size_t i = ((size_t)blockIdx.x * 256 + threadIdx.x) * 8;
  float4 a = ((const float4*)(in + i))[0];
  float4 b = ((const float4*)(in + i))[1];
  __bf16* o = out + i;
  o[0] = f2bf(a.x); o[1] = f2bf(a.y); o[2] = f2bf(a.z); o[3] = f2bf(a.w);
  o[4] = f2bf(b.x); o[5] = f2bf(b.y); o[6] = f2bf(b.z); o[7] = f2bf(b.w);
}

__global__ __launch_bounds__(256) void wtrans_kernel(
    const float* __restrict__ W, __bf16* __restrict__ Wt, int K, int N) {
  __shared__ float tile[32][33];
  const int kb = blockIdx.x * 32, nb = blockIdx.y * 32;
  const int tx = threadIdx.x & 31, ty = threadIdx.x >> 5;  // ty in 0..7
#pragma unroll
  for (int i = ty; i < 32; i += 8)
    tile[i][tx] = W[(size_t)(kb + i) * N + (nb + tx)];
  __syncthreads();
#pragma unroll
  for (int i = ty; i < 32; i += 8)
    Wt[(size_t)(nb + i) * K + (kb + tx)] = f2bf(tile[tx][i]);
}

// ---------------------------------------------------------------------------
// Host-side orchestration.
// Inputs: 0:x 1:mask(unused; causal structure known) 2:t_w 3:t_b 4:s_w 5:s_b
//         6:ff1_w 7:ff1_b 8:ff2_w 9:ff2_b
// ---------------------------------------------------------------------------
extern "C" void kernel_launch(void* const* d_in, const int* in_sizes, int n_in,
                              void* d_out, int out_size, void* d_ws, size_t ws_size,
                              hipStream_t stream) {
  (void)in_sizes; (void)n_in; (void)out_size; (void)ws_size;

  const float* x   = (const float*)d_in[0];
  const float* t_w = (const float*)d_in[2];
  const float* t_b = (const float*)d_in[3];
  const float* s_w = (const float*)d_in[4];
  const float* s_b = (const float*)d_in[5];
  const float* f1w = (const float*)d_in[6];
  const float* f1b = (const float*)d_in[7];
  const float* f2w = (const float*)d_in[8];
  const float* f2b = (const float*)d_in[9];
  float* out = (float*)d_out;

  const size_t WMAT = (size_t)D_MODEL * D_MODEL;
  char* ws = (char*)d_ws;
  __bf16* xb   = (__bf16*)ws; ws += (size_t)TOKENS * D_MODEL * 2;
  __bf16* tWt  = (__bf16*)ws; ws += 4 * WMAT * 2;               // [N][K] bf16
  __bf16* sWt  = (__bf16*)ws; ws += 4 * WMAT * 2;
  __bf16* f1Wt = (__bf16*)ws; ws += (size_t)D_MODEL * DFF * 2;  // [2048][512]
  __bf16* f2Wt = (__bf16*)ws; ws += (size_t)DFF * D_MODEL * 2;  // [512][2048]
  __bf16* Qb   = (__bf16*)ws; ws += (size_t)TOKENS * D_MODEL * 2;
  __bf16* Kb   = (__bf16*)ws; ws += (size_t)TOKENS * D_MODEL * 2;
  __bf16* Vb   = (__bf16*)ws; ws += (size_t)TOKENS * D_MODEL * 2;
  __bf16* AOb  = (__bf16*)ws; ws += (size_t)TOKENS * D_MODEL * 2;
  float*  Y1   = (float*)ws;  ws += (size_t)TOKENS * D_MODEL * 4;
  __bf16* Y1b  = (__bf16*)ws; ws += (size_t)TOKENS * D_MODEL * 2;
  float*  Y2   = (float*)ws;  ws += (size_t)TOKENS * D_MODEL * 4;
  __bf16* Y2b  = (__bf16*)ws; ws += (size_t)TOKENS * D_MODEL * 2;
  __bf16* Hb   = (__bf16*)ws; ws += (size_t)TOKENS * DFF * 2;

  const dim3 blk(256);
  const dim3 gproj(TOKENS / 128, D_MODEL / 128);  // 512 x 4
  const dim3 gff1(TOKENS / 128, DFF / 128);       // 512 x 16
  const dim3 gattn(NB * NHEADS * SSZ);            // 4096 head-slices
  const size_t attn_smem = (size_t)(3 * 128 * 64 * 2) + 128 * 128 * 4 + 128 * 128 * 2;

  // ---- prep: bf16 shadows of x and all weights (transposed to [N][K]) ----
  cvt_f32_bf16<<<(TOKENS * (size_t)D_MODEL) / (256 * 8), blk, 0, stream>>>(x, xb);
  for (int i = 0; i < 4; ++i) {
    wtrans_kernel<<<dim3(16, 16), blk, 0, stream>>>(t_w + i * WMAT, tWt + i * WMAT,
                                                    D_MODEL, D_MODEL);
    wtrans_kernel<<<dim3(16, 16), blk, 0, stream>>>(s_w + i * WMAT, sWt + i * WMAT,
                                                    D_MODEL, D_MODEL);
  }
  wtrans_kernel<<<dim3(16, 64), blk, 0, stream>>>(f1w, f1Wt, D_MODEL, DFF);
  wtrans_kernel<<<dim3(64, 16), blk, 0, stream>>>(f2w, f2Wt, DFF, D_MODEL);

  // ---- stage 1: time attention ----
  gemm_tdm_wmma<false, 0, false><<<gproj, blk, 0, stream>>>(
      xb, tWt + 0 * WMAT, t_b + 0 * D_MODEL, nullptr, nullptr, Qb,
      TOKENS, D_MODEL, D_MODEL);
  gemm_tdm_wmma<false, 0, false><<<gproj, blk, 0, stream>>>(
      xb, tWt + 1 * WMAT, t_b + 1 * D_MODEL, nullptr, nullptr, Kb,
      TOKENS, D_MODEL, D_MODEL);
  gemm_tdm_wmma<false, 0, false><<<gproj, blk, 0, stream>>>(
      xb, tWt + 2 * WMAT, t_b + 2 * D_MODEL, nullptr, nullptr, Vb,
      TOKENS, D_MODEL, D_MODEL);
  attn_wmma_kernel<<<gattn, blk, attn_smem, stream>>>(Qb, Kb, Vb, AOb, 0);
  gemm_tdm_wmma<false, 2, true><<<gproj, blk, 0, stream>>>(
      AOb, tWt + 3 * WMAT, t_b + 3 * D_MODEL, x, Y1, Y1b,
      TOKENS, D_MODEL, D_MODEL);

  // ---- stage 2: space attention ----
  gemm_tdm_wmma<false, 0, false><<<gproj, blk, 0, stream>>>(
      Y1b, sWt + 0 * WMAT, s_b + 0 * D_MODEL, nullptr, nullptr, Qb,
      TOKENS, D_MODEL, D_MODEL);
  gemm_tdm_wmma<false, 0, false><<<gproj, blk, 0, stream>>>(
      Y1b, sWt + 1 * WMAT, s_b + 1 * D_MODEL, nullptr, nullptr, Kb,
      TOKENS, D_MODEL, D_MODEL);
  gemm_tdm_wmma<false, 0, false><<<gproj, blk, 0, stream>>>(
      Y1b, sWt + 2 * WMAT, s_b + 2 * D_MODEL, nullptr, nullptr, Vb,
      TOKENS, D_MODEL, D_MODEL);
  attn_wmma_kernel<<<gattn, blk, attn_smem, stream>>>(Qb, Kb, Vb, AOb, 1);
  gemm_tdm_wmma<false, 2, true><<<gproj, blk, 0, stream>>>(
      AOb, sWt + 3 * WMAT, s_b + 3 * D_MODEL, Y1, Y2, Y2b,
      TOKENS, D_MODEL, D_MODEL);

  // ---- stage 3: FFN ----
  gemm_tdm_wmma<true, 0, false><<<gff1, blk, 0, stream>>>(
      Y2b, f1Wt, f1b, nullptr, nullptr, Hb, TOKENS, DFF, D_MODEL);
  gemm_tdm_wmma<false, 1, true><<<gproj, blk, 0, stream>>>(
      Hb, f2Wt, f2b, Y2, out, nullptr, TOKENS, D_MODEL, DFF);
}
